// MyScaledProductAttention_40252433498699
// MI455X (gfx1250) — matrix operand order repaired and checked by reading, not compiled, not run on hardware
//
#include <hip/hip_runtime.h>
#include <hip/hip_bf16.h>

// ---------------------------------------------------------------------------
// Flash attention for B=2, H=16, S=2048, D=128 (fp32 in/out) on gfx1250.
// bf16 WMMA (v_wmma_f32_16x16x32_bf16) for QK^T and PV, fp32 online softmax.
// Each wave owns 32 Q rows (2 M-tiles) so every LDS B-fragment feeds 2 WMMAs.
// ---------------------------------------------------------------------------

typedef __attribute__((ext_vector_type(16))) __bf16 v16bf;
typedef __attribute__((ext_vector_type(8)))  __bf16 bf16x8;
typedef __attribute__((ext_vector_type(8)))  float  v8f;

#define WMMA_BF16(a, b, c) \
  __builtin_amdgcn_wmma_f32_16x16x32_bf16(false, (a), false, (b), (short)0, (c), false, false)

namespace {
constexpr int kB  = 2;
constexpr int kH  = 16;
constexpr int kS  = 2048;
constexpr int kD  = 128;
constexpr int kBQ = 128;           // Q rows per block (4 waves x 32)
constexpr int kBK = 64;            // keys per tile
constexpr int kNBH = kB * kH;      // 32
constexpr int kTiles = kS / kBK;   // 32

constexpr int LDK_S = kD + 8;      // 136 halves per K row   (272 B, 16B multiple)
constexpr int LDV_S = kBK + 8;     // 72  halves per V^T row (144 B, 16B multiple)
constexpr int LDP_S = kBK + 8;     // 72  halves per P row
}

// Combine two 8x bf16 LDS chunks into one 16-wide fragment register set.
__device__ __forceinline__ v16bf ld_frag16(const __bf16* p_lo, const __bf16* p_hi) {
  bf16x8 lo = *(const bf16x8*)p_lo;
  bf16x8 hi = *(const bf16x8*)p_hi;
  v16bf r;
#pragma unroll
  for (int i = 0; i < 8; ++i) { r[i] = lo[i]; r[i + 8] = hi[i]; }
  return r;
}

__global__ __launch_bounds__(128) void fa_fwd_kernel(
    const float* __restrict__ Q, const float* __restrict__ K,
    const float* __restrict__ V, float* __restrict__ O) {
  __shared__ __align__(16) __bf16 ldsK[kBK * LDK_S];       // [key][d]
  __shared__ __align__(16) __bf16 ldsV[kD * LDV_S];        // [d][key]  (transposed)
  __shared__ __align__(16) __bf16 ldsP[4][32 * LDP_S];     // per-wave [qrow 0..31][key]

  const int bh  = blockIdx.y;
  const int q0  = blockIdx.x * kBQ;
  const int tid = threadIdx.x;
  const int w   = tid >> 5;          // wave id 0..3
  const int lane = tid & 31;
  const int l16  = lane & 15;
  const int hi   = lane >> 4;        // 0 for lanes 0-15, 1 for 16-31

  const size_t base = (size_t)bh * kS * kD;
  const float* qp = Q + base;
  const float* kp = K + base;
  const float* vp = V + base;

  // fold 1/sqrt(D) and log2(e) into Q so softmax uses exp2
  const float qscale = 0.08838834764831845f * 1.4426950408889634f;

  // Staging index maps (per thread)
  const int krow_i = tid >> 1;            // 0..63   key row for K staging
  const int kcol_i = (tid & 1) * 64;      // 0 / 64  d half for K staging
  const int vkey_i = (tid & 7) * 8;       // 0..56   key base for V staging (8x8 block)
  const int vdim_i = (tid >> 3) * 8;      // 0..120  d base for V staging

  // --- Load Q fragments (A layout: 16x32 bf16 per k-step, 4 k-steps for D=128)
  // lanes 0-15: row l16, halves = K 0..7 then 16..23 of the k-step;
  // lanes 16-31: row l16, halves = K 8..15 then 24..31.
  v16bf a_q[2][4];
#pragma unroll
  for (int mt = 0; mt < 2; ++mt) {
    const float* qr = qp + (size_t)(q0 + w * 32 + mt * 16 + l16) * kD;
#pragma unroll
    for (int k = 0; k < 4; ++k) {
      const int c0 = k * 32 + hi * 8;
      const int c1 = c0 + 16;
      alignas(16) float f0[8], f1[8];
      *(float4*)&f0[0] = *(const float4*)(qr + c0);
      *(float4*)&f0[4] = *(const float4*)(qr + c0 + 4);
      *(float4*)&f1[0] = *(const float4*)(qr + c1);
      *(float4*)&f1[4] = *(const float4*)(qr + c1 + 4);
#pragma unroll
      for (int i = 0; i < 8; ++i) {
        a_q[mt][k][i]     = (__bf16)(f0[i] * qscale);
        a_q[mt][k][i + 8] = (__bf16)(f1[i] * qscale);
      }
    }
  }

  // --- Accumulators: O is 2x(16x128) fp32 = 2x8 C-fragments; m/l per row
  v8f o_acc[2][8];
#pragma unroll
  for (int mt = 0; mt < 2; ++mt)
#pragma unroll
    for (int i = 0; i < 8; ++i)
#pragma unroll
      for (int j = 0; j < 8; ++j) o_acc[mt][i][j] = 0.0f;
  float m_r[2][8], l_r[2][8];
#pragma unroll
  for (int mt = 0; mt < 2; ++mt)
#pragma unroll
    for (int r = 0; r < 8; ++r) { m_r[mt][r] = -__builtin_inff(); l_r[mt][r] = 0.0f; }

  // =============================== key tiles ===============================
  for (int kt = 0; kt < kTiles; ++kt) {
    const int key0 = kt * kBK;

    // --- Cooperative stage: K row-major bf16 (vector stores)
    {
      const float* krow = kp + (size_t)(key0 + krow_i) * kD + kcol_i;
#pragma unroll
      for (int c = 0; c < 64; c += 8) {
        alignas(16) float f[8];
        *(float4*)&f[0] = *(const float4*)(krow + c);
        *(float4*)&f[4] = *(const float4*)(krow + c + 4);
        bf16x8 hk;
#pragma unroll
        for (int i = 0; i < 8; ++i) hk[i] = (__bf16)f[i];
        *(bf16x8*)&ldsK[krow_i * LDK_S + kcol_i + c] = hk;
      }
    }
    // --- Cooperative stage: V transposed bf16 via 8x8 register transpose
    //     (vector stores: 8 ds_store_b128-class ops instead of 64 scalar b16)
    {
#pragma unroll
      for (int h = 0; h < 2; ++h) {
        alignas(16) float fv[8][4];
#pragma unroll
        for (int i = 0; i < 8; ++i)
          *(float4*)&fv[i][0] =
              *(const float4*)(vp + (size_t)(key0 + vkey_i + i) * kD + vdim_i + h * 4);
#pragma unroll
        for (int j = 0; j < 4; ++j) {
          bf16x8 hv;
#pragma unroll
          for (int i = 0; i < 8; ++i) hv[i] = (__bf16)fv[i][j];
          *(bf16x8*)&ldsV[(vdim_i + h * 4 + j) * LDV_S + vkey_i] = hv;
        }
      }
    }
    __syncthreads();

    // Prefetch next tile's K/V rows into caches (global_prefetch_b8)
    if (kt + 1 < kTiles) {
      const int nkey0 = key0 + kBK;
      __builtin_prefetch(kp + (size_t)(nkey0 + krow_i) * kD + kcol_i, 0, 3);
      __builtin_prefetch(vp + (size_t)(nkey0 + vkey_i) * kD + vdim_i, 0, 3);
    }

    // --- S = Q * K^T : 4 N-tiles of 16 keys; burst-load 4 K-step fragments,
    //     then 8 back-to-back WMMAs (4 k-steps x 2 M-tiles share each B frag).
    v8f s_acc[2][4];
#pragma unroll
    for (int nt = 0; nt < 4; ++nt) {
      v16bf bk[4];
#pragma unroll
      for (int k = 0; k < 4; ++k) {
        const __bf16* kr = &ldsK[(nt * 16 + l16) * LDK_S + k * 32 + hi * 8];
        bk[k] = ld_frag16(kr, kr + 16);
      }
#pragma unroll
      for (int mt = 0; mt < 2; ++mt)
#pragma unroll
        for (int j = 0; j < 8; ++j) s_acc[mt][nt][j] = 0.0f;
#pragma unroll
      for (int k = 0; k < 4; ++k) {
        s_acc[0][nt] = WMMA_BF16(a_q[0][k], bk[k], s_acc[0][nt]);
        s_acc[1][nt] = WMMA_BF16(a_q[1][k], bk[k], s_acc[1][nt]);
      }
    }

    // --- Online softmax (rows: r for lanes 0-15, r+8 for lanes 16-31)
    float alpha[2][8];
#pragma unroll
    for (int mt = 0; mt < 2; ++mt) {
      float mnew[8];
#pragma unroll
      for (int r = 0; r < 8; ++r) {
        float t = fmaxf(fmaxf(s_acc[mt][0][r], s_acc[mt][1][r]),
                        fmaxf(s_acc[mt][2][r], s_acc[mt][3][r]));
#pragma unroll
        for (int msk = 1; msk < 16; msk <<= 1) t = fmaxf(t, __shfl_xor(t, msk, 32));
        mnew[r]      = fmaxf(m_r[mt][r], t);
        alpha[mt][r] = exp2f(m_r[mt][r] - mnew[r]);  // 0 on first tile (m=-inf)
        m_r[mt][r]   = mnew[r];
      }
#pragma unroll
      for (int r = 0; r < 8; ++r) {
        float sum = 0.0f;
#pragma unroll
        for (int nt = 0; nt < 4; ++nt) {
          float p = exp2f(s_acc[mt][nt][r] - mnew[r]);
          s_acc[mt][nt][r] = p;
          sum += p;
        }
#pragma unroll
        for (int msk = 1; msk < 16; msk <<= 1) sum += __shfl_xor(sum, msk, 32);
        l_r[mt][r] = l_r[mt][r] * alpha[mt][r] + sum;
      }
#pragma unroll
      for (int dt = 0; dt < 8; ++dt)
#pragma unroll
        for (int r = 0; r < 8; ++r) o_acc[mt][dt][r] *= alpha[mt][r];
    }

    // --- P (C layout) -> wave-private LDS -> A fragments
    __bf16* pw = &ldsP[w][0];
#pragma unroll
    for (int mt = 0; mt < 2; ++mt)
#pragma unroll
      for (int nt = 0; nt < 4; ++nt)
#pragma unroll
        for (int r = 0; r < 8; ++r)
          pw[(mt * 16 + r + hi * 8) * LDP_S + nt * 16 + l16] =
              (__bf16)s_acc[mt][nt][r];
    // DS ops of one wave are in-order; only the counter wait is needed.
    asm volatile("s_wait_dscnt 0" ::: "memory");

    v16bf p_a[2][2];
#pragma unroll
    for (int mt = 0; mt < 2; ++mt)
#pragma unroll
      for (int ks = 0; ks < 2; ++ks) {
        const __bf16* pr = &pw[(mt * 16 + l16) * LDP_S + ks * 32 + hi * 8];
        p_a[mt][ks] = ld_frag16(pr, pr + 16);
      }

    // --- O += P * V : 8 N-tiles of 16 dims; each V frag feeds 2 WMMAs
#pragma unroll
    for (int dt = 0; dt < 8; ++dt) {
      v16bf bv[2];
#pragma unroll
      for (int ks = 0; ks < 2; ++ks) {
        const __bf16* vr = &ldsV[(dt * 16 + l16) * LDV_S + ks * 32 + hi * 8];
        bv[ks] = ld_frag16(vr, vr + 16);
      }
#pragma unroll
      for (int ks = 0; ks < 2; ++ks) {
        o_acc[0][dt] = WMMA_BF16(p_a[0][ks], bv[ks], o_acc[0][dt]);
        o_acc[1][dt] = WMMA_BF16(p_a[1][ks], bv[ks], o_acc[1][dt]);
      }
    }
    __syncthreads();  // protect ldsK/ldsV before next stage
  }

  // --- Normalize by l and write fp32 output
  float* op = O + base;
#pragma unroll
  for (int mt = 0; mt < 2; ++mt) {
    const int orow0 = q0 + w * 32 + mt * 16 + hi * 8;
#pragma unroll
    for (int r = 0; r < 8; ++r) {
      const float inv = 1.0f / l_r[mt][r];
      float* orow = op + (size_t)(orow0 + r) * kD + l16;
#pragma unroll
      for (int dt = 0; dt < 8; ++dt) orow[dt * 16] = o_acc[mt][dt][r] * inv;
    }
  }
}

extern "C" void kernel_launch(void* const* d_in, const int* in_sizes, int n_in,
                              void* d_out, int out_size, void* d_ws, size_t ws_size,
                              hipStream_t stream) {
  (void)in_sizes; (void)n_in; (void)out_size; (void)d_ws; (void)ws_size;
  const float* q = (const float*)d_in[0];
  const float* k = (const float*)d_in[1];
  const float* v = (const float*)d_in[2];
  float* out = (float*)d_out;

  dim3 grid(kS / kBQ, kNBH);   // (16, 32)
  dim3 block(128);
  fa_fwd_kernel<<<grid, block, 0, stream>>>(q, k, v, out);
}
